// GatedDeltaNet_79173427135023
// MI455X (gfx1250) — matrix operand order repaired and checked by reading, not compiled, not run on hardware
//
#include <hip/hip_runtime.h>
#include <hip/hip_bf16.h>
#include <stdint.h>

// ---------------- problem constants (from reference setup_inputs) ------------
#define T_LEN   1024
#define HIDN    2048
#define NHEADS  12
#define DKH     128
#define DVH     256
#define KDIM    (NHEADS * DKH)   // 1536
#define VDIM    (NHEADS * DVH)   // 3072

typedef __attribute__((ext_vector_type(16))) __bf16 v16bf;
typedef __attribute__((ext_vector_type(8)))  float  v8f;

// ---------------- helpers ----------------------------------------------------
__device__ __forceinline__ uint16_t f2bf(float f) {
  union { float f; uint32_t u; } x; x.f = f;
  uint32_t u = x.u;
  u += 0x7FFFu + ((u >> 16) & 1u);          // round-to-nearest-even
  return (uint16_t)(u >> 16);
}
__device__ __forceinline__ float sigmoidf_(float x) { return 1.f / (1.f + __expf(-x)); }
__device__ __forceinline__ float siluf_(float x)    { return x / (1.f + __expf(-x)); }

union FragU { uint4 q[2]; v16bf v; };

// A fragment from an LDS-staged 32x32 bf16 tile (row-major, ld=32):
//   lane half h, local row r: e0..15 -> K = 16h+0..15 (contiguous)
__device__ __forceinline__ v16bf load_a_frag_lds(const uint16_t* tile, int rowl, int half) {
  const uint16_t* p = tile + rowl * 32 + half * 16;
  FragU u;
  u.q[0] = *reinterpret_cast<const uint4*>(p);        // ds_load_b128
  u.q[1] = *reinterpret_cast<const uint4*>(p + 8);    // ds_load_b128
  return u.v;
}
// B fragment, 32x16 bf16 with B pre-transposed to Bt[N,K]:
//   lane half h, col r: elements e0..15 -> K = k0 + 16h + 0..15 (contiguous)
__device__ __forceinline__ v16bf load_b_frag(const uint16_t* __restrict__ Bt, int ldk,
                                             int colN, int k0, int half) {
  const uint16_t* p = Bt + (size_t)colN * ldk + k0 + half * 16;
  FragU u;
  u.q[0] = *reinterpret_cast<const uint4*>(p);
  u.q[1] = *reinterpret_cast<const uint4*>(p + 8);
  return u.v;
}

// ---------------- kernel: f32 -> bf16 ----------------------------------------
__global__ void to_bf16_kernel(const float* __restrict__ src, uint16_t* __restrict__ dst, int n) {
  int i = blockIdx.x * blockDim.x + threadIdx.x;
  if (i < n) dst[i] = f2bf(src[i]);
}

// ---------------- kernel: transpose + convert  src[R,C] f32 -> dst[C,R] bf16 -
__global__ void transp_bf16_kernel(const float* __restrict__ src, uint16_t* __restrict__ dst,
                                   int R, int C) {
  __shared__ float tile[32][33];
  int c0 = blockIdx.x * 32 + threadIdx.x;
  int r0 = blockIdx.y * 32 + threadIdx.y;
#pragma unroll
  for (int j = 0; j < 4; ++j)
    tile[threadIdx.y + 8 * j][threadIdx.x] = src[(size_t)(r0 + 8 * j) * C + c0];
  __syncthreads();
  int c2 = blockIdx.x * 32 + threadIdx.y;
  int r2 = blockIdx.y * 32 + threadIdx.x;
#pragma unroll
  for (int j = 0; j < 4; ++j)
    dst[(size_t)(c2 + 8 * j) * R + r2] = f2bf(tile[threadIdx.x][threadIdx.y + 8 * j]);
}

// ---------------- kernel: bf16 WMMA GEMM  C[M,N] = A[M,K] * Bt[N,K]^T --------
// grid = (tilesN/8, M/32); block = 256 (8 waves). All waves share the 32-row
// M-strip; the A tile is staged into LDS with the gfx1250 async Global->LDS
// engine (ASYNCcnt), 3-deep pipelined: issue(j+2) is placed after compute(j)
// so a single barrier per K-step gives both visibility and overwrite safety.
// Steady-state inner loop is branch-free with literal s_wait_asynccnt 1.
__global__ void gemm_bf16_wmma(const uint16_t* __restrict__ A,
                               const uint16_t* __restrict__ Bt,
                               float* __restrict__ C, int M, int N, int K) {
  __shared__ __align__(16) uint16_t tileA[3][32 * 32];   // 3 x 2KB

  const int tid  = threadIdx.x;
  const int lane = tid & 31;
  const int wid  = tid >> 5;
  const int m0   = blockIdx.y << 5;
  const int n0   = (blockIdx.x * 8 + wid) << 5;
  const int half = lane >> 4;
  const int r    = lane & 15;

  // async staging: 256 threads x 8B = 2KB tile, 1 async instr per wave per stage
  const int srow = tid >> 3;        // 0..31
  const int sseg = tid & 7;         // 8B units within a 64B row

  auto stage = [&](int buf, int k0) {
    uint32_t ldsa = (uint32_t)(uintptr_t)&tileA[buf][srow * 32 + sseg * 4];
    uint64_t ga   = (uint64_t)(uintptr_t)(A + (size_t)(m0 + srow) * K + k0 + sseg * 4);
    asm volatile("global_load_async_to_lds_b64 %0, %1, off" :: "v"(ldsa), "v"(ga) : "memory");
  };

  v8f c00 = {}, c01 = {}, c10 = {}, c11 = {};
  auto compute = [&](int buf, int k0) {
    v16bf a0 = load_a_frag_lds(&tileA[buf][0], r,      half);
    v16bf a1 = load_a_frag_lds(&tileA[buf][0], 16 + r, half);
    v16bf b0 = load_b_frag(Bt, K, n0 + r,      k0, half);
    v16bf b1 = load_b_frag(Bt, K, n0 + 16 + r, k0, half);
    c00 = __builtin_amdgcn_wmma_f32_16x16x32_bf16(false, a0, false, b0, (short)0, c00, false, false);
    c01 = __builtin_amdgcn_wmma_f32_16x16x32_bf16(false, a0, false, b1, (short)0, c01, false, false);
    c10 = __builtin_amdgcn_wmma_f32_16x16x32_bf16(false, a1, false, b0, (short)0, c10, false, false);
    c11 = __builtin_amdgcn_wmma_f32_16x16x32_bf16(false, a1, false, b1, (short)0, c11, false, false);
  };

  const int ns = K >> 5;            // #K-stages (64 or 96 here, always >= 3)
  stage(0, 0);
  stage(1, 32);

  int bc = 0;                        // compute buffer, rotates 0,1,2
  int k0 = 0;
  for (int j = 0; j < ns - 2; ++j, k0 += 32) {
    asm volatile("s_wait_asynccnt 1" ::: "memory");   // stage j complete (own issues)
    __syncthreads();                                  // ... for all waves
    compute(bc, k0);
    // prefetch the B strip two stages ahead (global_prefetch_b8)
    __builtin_prefetch(Bt + (size_t)(n0 + r) * K + k0 + 64, 0, 3);
    // issue stage j+2: its target buffer was last read at iteration j-1 and
    // every wave passed barrier(j) after compute(j-1) -> overwrite-safe.
    int bs = bc + 2; if (bs >= 3) bs -= 3;
    stage(bs, k0 + 64);
    if (++bc == 3) bc = 0;
  }
  // epilogue (2 peeled iterations; constant-immediate waits)
  asm volatile("s_wait_asynccnt 1" ::: "memory");
  __syncthreads();
  compute(bc, k0);
  if (++bc == 3) bc = 0;
  k0 += 32;
  asm volatile("s_wait_asynccnt 0" ::: "memory");
  __syncthreads();
  compute(bc, k0);

  // C/D layout: lanes 0-15 N=lane,M=vgpr i; lanes 16-31 N=lane-16,M=8+i
#pragma unroll
  for (int i = 0; i < 8; ++i) {
    int mA = m0 + half * 8 + i;
    int mB = m0 + 16 + half * 8 + i;
    C[(size_t)mA * N + n0 + r]      = c00[i];
    C[(size_t)mA * N + n0 + 16 + r] = c01[i];
    C[(size_t)mB * N + n0 + r]      = c10[i];
    C[(size_t)mB * N + n0 + 16 + r] = c11[i];
  }
}

// ---------------- kernel: beta / g projections (N=12 each) -------------------
__global__ void proj_beta_g(const float* __restrict__ h, const float* __restrict__ Wb,
                            const float* __restrict__ Wa, const float* __restrict__ A_log,
                            const float* __restrict__ dt_bias,
                            float* __restrict__ beta, float* __restrict__ g) {
  int idx = blockIdx.x * blockDim.x + threadIdx.x;
  if (idx >= T_LEN * NHEADS) return;
  int hh = idx % NHEADS;
  int t  = idx / NHEADS;
  const float* hr = h + (size_t)t * HIDN;
  float ab = 0.f, aa = 0.f;
  for (int k = 0; k < HIDN; ++k) {
    float hv = hr[k];
    ab += hv * Wb[k * NHEADS + hh];
    aa += hv * Wa[k * NHEADS + hh];
  }
  beta[idx] = sigmoidf_(ab);
  float xv = aa + dt_bias[hh];
  float sp = (xv > 20.f) ? xv : log1pf(__expf(xv));   // softplus
  g[idx] = -__expf(A_log[hh]) * sp;
}

// ---------------- kernel: causal depthwise conv(K=4) + SiLU (+ l2norm) ------
template <int CH, bool NORM>
__global__ void conv_silu_kernel(const float* __restrict__ x, const float* __restrict__ w,
                                 float* __restrict__ y, int C) {
  int b  = blockIdx.x;
  int nh = C / CH;
  int t  = b / nh;
  int hh = b - t * nh;
  int c  = hh * CH + threadIdx.x;
  float acc = 0.f;
#pragma unroll
  for (int i = 0; i < 4; ++i) {
    int tt = t - 3 + i;
    if (tt >= 0) acc += x[(size_t)tt * C + c] * w[c * 4 + i];
  }
  float s = siluf_(acc);
  if (NORM) {
    __shared__ float red[CH];
    red[threadIdx.x] = s * s;
    __syncthreads();
    for (int off = CH / 2; off > 0; off >>= 1) {
      if (threadIdx.x < off) red[threadIdx.x] += red[threadIdx.x + off];
      __syncthreads();
    }
    s *= rsqrtf(red[0] + 1e-6f);
  }
  y[(size_t)t * C + c] = s;
}

// ---------------- kernel: gated delta rule scan ------------------------------
// block = (head, dv-slice of 64); S[128,64] resident in LDS (padded vs bank conflicts)
__global__ void delta_scan(const float* __restrict__ q, const float* __restrict__ k,
                           const float* __restrict__ v, const float* __restrict__ g,
                           const float* __restrict__ beta, float* __restrict__ o) {
  const int head = blockIdx.x >> 2;
  const int vs0  = (blockIdx.x & 3) * 64;
  const int tid  = threadIdx.x;              // 256 threads
  const int col  = tid & 63;
  const int i0   = (tid >> 6) * 32;          // DK quarter

  __shared__ float S[128 * 65];              // padded: bank = (i+col)%64
  __shared__ float sk[DKH], sq[DKH], svt[64], red[256], sdelta[64], sgb[2];

  for (int idx = tid; idx < 128 * 65; idx += 256) S[idx] = 0.f;
  __syncthreads();

  for (int t = 0; t < T_LEN; ++t) {
    const int base = t * NHEADS + head;
    if (tid < 128) {
      sk[tid] = k[(size_t)base * DKH + tid];
      sq[tid] = q[(size_t)base * DKH + tid];
    } else if (tid < 192) {
      svt[tid - 128] = v[(size_t)base * DVH + vs0 + (tid - 128)];
    } else if (tid == 192) {
      sgb[0] = __expf(g[base]);
      sgb[1] = beta[base];
    }
    __syncthreads();
    const float dec = sgb[0], bt = sgb[1];

    // partial k^T S over this thread's DK quarter
    float p = 0.f;
#pragma unroll 8
    for (int i = 0; i < 32; ++i) p += sk[i0 + i] * S[(i0 + i) * 65 + col];
    red[tid] = p;
    __syncthreads();
    if (tid < 64) {
      float kd = red[col] + red[col + 64] + red[col + 128] + red[col + 192];
      sdelta[col] = (svt[col] - dec * kd) * bt;   // delta-rule error * beta
    }
    __syncthreads();

    // S = S*exp(g) + k (x) delta, fused with partial q^T S_new
    const float d = sdelta[col];
    float qp = 0.f;
#pragma unroll 8
    for (int i = 0; i < 32; ++i) {
      int ii = i0 + i;
      float s = S[ii * 65 + col] * dec + sk[ii] * d;
      S[ii * 65 + col] = s;
      qp += sq[ii] * s;
    }
    red[tid] = qp;
    __syncthreads();
    if (tid < 64)
      o[(size_t)base * DVH + vs0 + col] =
          red[col] + red[col + 64] + red[col + 128] + red[col + 192];
    __syncthreads();
  }
}

// ---------------- kernel: RMSNorm * w * SiLU(gate) -> bf16 -------------------
__global__ void rmsnorm_gate(const float* __restrict__ o, const float* __restrict__ gg,
                             const float* __restrict__ norm_w, uint16_t* __restrict__ onbf) {
  int b  = blockIdx.x;
  int t  = b / NHEADS;
  int hh = b - t * NHEADS;
  int c  = threadIdx.x;                     // 256 threads
  float ov = o[((size_t)t * NHEADS + hh) * DVH + c];
  __shared__ float red[256];
  red[c] = ov * ov;
  __syncthreads();
  for (int off = 128; off > 0; off >>= 1) {
    if (c < off) red[c] += red[c + off];
    __syncthreads();
  }
  float r  = rsqrtf(red[0] * (1.f / DVH) + 1e-5f);
  float gv = gg[(size_t)t * VDIM + hh * DVH + c];
  float res = ov * r * norm_w[c] * siluf_(gv);
  onbf[(size_t)t * VDIM + hh * DVH + c] = f2bf(res);
}

// ---------------- host orchestration -----------------------------------------
extern "C" void kernel_launch(void* const* d_in, const int* in_sizes, int n_in,
                              void* d_out, int out_size, void* d_ws, size_t ws_size,
                              hipStream_t stream) {
  const float* hs      = (const float*)d_in[0];
  const float* Wq      = (const float*)d_in[1];
  const float* Wk      = (const float*)d_in[2];
  const float* Wv      = (const float*)d_in[3];
  const float* Wb      = (const float*)d_in[4];
  const float* Wa      = (const float*)d_in[5];
  const float* Wg      = (const float*)d_in[6];
  const float* Wo      = (const float*)d_in[7];
  const float* cwq     = (const float*)d_in[8];
  const float* cwk     = (const float*)d_in[9];
  const float* cwv     = (const float*)d_in[10];
  const float* A_log   = (const float*)d_in[11];
  const float* dt_bias = (const float*)d_in[12];
  const float* norm_w  = (const float*)d_in[13];
  float* outF = (float*)d_out;

  uint8_t* wsp = (uint8_t*)d_ws;
  auto carve = [&](size_t bytes) -> void* {
    void* p = (void*)wsp;
    wsp += (bytes + 255) & ~(size_t)255;
    return p;
  };
  uint16_t* hbf  = (uint16_t*)carve((size_t)T_LEN * HIDN * 2);
  uint16_t* wt   = (uint16_t*)carve((size_t)VDIM * HIDN * 2);   // reused per GEMM
  float*    qpre = (float*)carve((size_t)T_LEN * KDIM * 4);
  float*    kpre = (float*)carve((size_t)T_LEN * KDIM * 4);
  float*    vpre = (float*)carve((size_t)T_LEN * VDIM * 4);
  float*    gg   = (float*)carve((size_t)T_LEN * VDIM * 4);
  float*    qn   = (float*)carve((size_t)T_LEN * KDIM * 4);
  float*    kn   = (float*)carve((size_t)T_LEN * KDIM * 4);
  float*    vs   = (float*)carve((size_t)T_LEN * VDIM * 4);
  float*    betaB= (float*)carve((size_t)T_LEN * NHEADS * 4);
  float*    gB   = (float*)carve((size_t)T_LEN * NHEADS * 4);
  float*    o    = vpre;               // alias: vpre dead after conv_v
  uint16_t* onbf = (uint16_t*)qpre;    // alias: qpre dead after conv_q

  // 1) hidden -> bf16
  {
    int n = T_LEN * HIDN;
    to_bf16_kernel<<<(n + 255) / 256, 256, 0, stream>>>(hs, hbf, n);
  }
  // 2-5) projection GEMMs (transpose-convert weight, then WMMA GEMM)
  transp_bf16_kernel<<<dim3(KDIM / 32, HIDN / 32), dim3(32, 8), 0, stream>>>(Wq, wt, HIDN, KDIM);
  gemm_bf16_wmma<<<dim3((KDIM / 32) / 8, T_LEN / 32), 256, 0, stream>>>(hbf, wt, qpre, T_LEN, KDIM, HIDN);
  transp_bf16_kernel<<<dim3(KDIM / 32, HIDN / 32), dim3(32, 8), 0, stream>>>(Wk, wt, HIDN, KDIM);
  gemm_bf16_wmma<<<dim3((KDIM / 32) / 8, T_LEN / 32), 256, 0, stream>>>(hbf, wt, kpre, T_LEN, KDIM, HIDN);
  transp_bf16_kernel<<<dim3(VDIM / 32, HIDN / 32), dim3(32, 8), 0, stream>>>(Wv, wt, HIDN, VDIM);
  gemm_bf16_wmma<<<dim3((VDIM / 32) / 8, T_LEN / 32), 256, 0, stream>>>(hbf, wt, vpre, T_LEN, VDIM, HIDN);
  transp_bf16_kernel<<<dim3(VDIM / 32, HIDN / 32), dim3(32, 8), 0, stream>>>(Wg, wt, HIDN, VDIM);
  gemm_bf16_wmma<<<dim3((VDIM / 32) / 8, T_LEN / 32), 256, 0, stream>>>(hbf, wt, gg, T_LEN, VDIM, HIDN);
  // 6) beta / g (fp32, tiny N)
  proj_beta_g<<<(T_LEN * NHEADS + 255) / 256, 256, 0, stream>>>(hs, Wb, Wa, A_log, dt_bias, betaB, gB);
  // 7) short conv + SiLU (+ per-head l2norm for q,k)
  conv_silu_kernel<128, true ><<<T_LEN * NHEADS, 128, 0, stream>>>(qpre, cwq, qn, KDIM);
  conv_silu_kernel<128, true ><<<T_LEN * NHEADS, 128, 0, stream>>>(kpre, cwk, kn, KDIM);
  conv_silu_kernel<256, false><<<T_LEN * NHEADS, 256, 0, stream>>>(vpre, cwv, vs, VDIM);
  // 8) gated delta rule scan (LDS-resident state, 48 blocks)
  delta_scan<<<NHEADS * 4, 256, 0, stream>>>(qn, kn, vs, gB, betaB, o);
  // 9) fused RMSNorm * norm_w * SiLU(gate) -> bf16
  rmsnorm_gate<<<T_LEN * NHEADS, 256, 0, stream>>>(o, gg, norm_w, onbf);
  // 10) output GEMM: [1024,3072] x Wo[3072,2048] -> d_out (f32)
  transp_bf16_kernel<<<dim3(HIDN / 32, VDIM / 32), dim3(32, 8), 0, stream>>>(Wo, wt, VDIM, HIDN);
  gemm_bf16_wmma<<<dim3((HIDN / 32) / 8, T_LEN / 32), 256, 0, stream>>>(onbf, wt, outF, T_LEN, HIDN, VDIM);
}
